// CausalLMWrapperBase_53068615910223
// MI455X (gfx1250) — matrix operand order repaired and checked by reading, not compiled, not run on hardware
//
#include <hip/hip_runtime.h>
#include <hip/hip_bf16.h>
#include <math.h>

typedef __attribute__((ext_vector_type(16))) __bf16 v16bf;
typedef __attribute__((ext_vector_type(8)))  float  v8f;

#define N_TOK   2047      // S-1 valid token positions
#define H_DIM   2048
#define V_DIM   32000
#define NVCHUNK 25        // vocab chunks
#define VCHUNK  1280      // V_DIM / NVCHUNK
#define MTILES16 128      // 16-token tiles (fallback)
#define MTILES64 32       // 64-token tiles (fast path)
#define LDS_UINTS (64*32*8)        // fallback path: 16384 uints = 64 KB
#define TILE_UINTS 16384           // one 16-row A tile = 64 ks * 32 lanes * 8 uints
#define FAST_LDS_BYTES (4*TILE_UINTS*4)   // 256 KB: 4 A tiles staged in LDS

// ---------- workspace layout (byte offsets) ----------
// [0, 8192)            : ws_label  (2048 f32, atomically accumulated)
// [8192, 417792)       : ws_ms     (2048 * 25 * 2 f32)
// fast path only:
// [417792, 8806400)    : a_pack    (128 tiles * 16384 uints, bf16 WMMA-A layout)
// [8806400, 139878400) : wbf       (32000 * 2048 bf16 weight, row-major)
#define OFF_MS    8192
#define OFF_APACK 417792
#define OFF_WBF   8806400
#define WS_NEED_FAST 139878400ull

struct U32x8 { uint4 lo; uint4 hi; };   // 32 bytes, 16B-aligned loads

__device__ inline unsigned int pk_bf16x2(float a, float b) {
    unsigned short u0 = __builtin_bit_cast(unsigned short, (__bf16)a);
    unsigned short u1 = __builtin_bit_cast(unsigned short, (__bf16)b);
    return (unsigned int)u0 | ((unsigned int)u1 << 16);
}

__global__ void ce_init_kernel(float* ws_label) {
    for (int i = threadIdx.x; i < 2048; i += blockDim.x) ws_label[i] = 0.0f;
}

// ---------- fast path: one-time fp32 -> bf16 weight conversion (131 MB, L2-resident) ----------
__global__ __launch_bounds__(256)
void ce_cvtw_kernel(const float* __restrict__ w, uint2* __restrict__ wbf) {
    size_t gid = (size_t)blockIdx.x * 256 + threadIdx.x;   // one float4 each
    const float4* s = reinterpret_cast<const float4*>(w);
    float4 f = s[gid];
    wbf[gid] = make_uint2(pk_bf16x2(f.x, f.y), pk_bf16x2(f.z, f.w));
}

// ---------- fast path: pre-pack hidden states into WMMA A-register layout (8 MB bf16) ----------
// a_pack[t16][ks][lane][j] : uint = bf16 pair, 16-bit A layout of a 16x32 tile
__global__ __launch_bounds__(256)
void ce_packa_kernel(const float* __restrict__ hs, unsigned int* __restrict__ a_pack) {
    int gid   = blockIdx.x * 256 + threadIdx.x;   // 2,097,152 uints total
    int j     = gid & 7;
    int lane  = (gid >> 3) & 31;
    int ks    = (gid >> 8) & 63;
    int t16   = gid >> 14;                        // 16-row tile index 0..127
    int row   = lane & 15;
    int khalf = lane >> 4;
    int e     = 2 * j;
    int kl    = e + khalf * 16;
    int k     = ks * 32 + kl;
    int n     = t16 * 16 + row;
    float x0 = 0.0f, x1 = 0.0f;
    if (n < N_TOK) {
        x0 = hs[(size_t)n * H_DIM + k];
        x1 = hs[(size_t)n * H_DIM + k + 1];
    }
    a_pack[gid] = pk_bf16x2(x0, x1);
}

// ---------- fast pass 1: M=64 per block, A staged once in 256 KB LDS, B reused by 4 WMMAs ----------
__global__ __launch_bounds__(256)
void ce_pass1_fast(const int* __restrict__ labels,
                   const unsigned short* __restrict__ wbf,
                   const unsigned int* __restrict__ a_pack,
                   float* __restrict__ ws_ms,
                   float* __restrict__ ws_label)
{
    extern __shared__ unsigned int ldsA[];   // 4 * 16384 uints = 256 KB

    const int mb  = blockIdx.x;    // 64-token tile, 0..31
    const int vc  = blockIdx.y;
    const int tid = threadIdx.x;

    // ---- stage pre-packed A slice (contiguous 64K uints) into LDS, once ----
    {
        const uint4* src = reinterpret_cast<const uint4*>(a_pack + (size_t)mb * 4 * TILE_UINTS);
        uint4*       dst = reinterpret_cast<uint4*>(ldsA);
        for (int i = tid; i < TILE_UINTS; i += 256)   // 16384 uint4 = 256 KB
            dst[i] = src[i];
    }
    __syncthreads();

    const int lane   = tid & 31;
    const int waveId = tid >> 5;
    const int col    = lane & 15;
    const int khalf  = lane >> 4;

    float rm[32], rs[32];
    #pragma unroll
    for (int r = 0; r < 32; ++r) { rm[r] = -INFINITY; rs[r] = 0.0f; }

    for (int vt = 0; vt < VCHUNK / 128; ++vt) {
        const int v = vc * VCHUNK + vt * 128 + waveId * 16 + col;  // vocab row
        const unsigned short* wrow = wbf + (size_t)v * H_DIM + khalf * 16;

        v8f cc[4];
        #pragma unroll
        for (int t = 0; t < 4; ++t) cc[t] = (v8f){};

        #pragma unroll 2
        for (int ks = 0; ks < H_DIM / 32; ++ks) {
            // B: 2x global_load_b128 of bf16 (16B-aligned uint4 pair)
            const uint4* bp = reinterpret_cast<const uint4*>(wrow + ks * 32);
            U32x8 bu; bu.lo = bp[0]; bu.hi = bp[1];
            v16bf b = __builtin_bit_cast(v16bf, bu);
            #pragma unroll
            for (int t = 0; t < 4; ++t) {
                // A: 2x ds_load_b128 (16B-aligned uint4 pair)
                const uint4* ap = reinterpret_cast<const uint4*>(
                    &ldsA[t * TILE_UINTS + ks * 256 + lane * 8]);
                U32x8 au; au.lo = ap[0]; au.hi = ap[1];
                v16bf a = __builtin_bit_cast(v16bf, au);
                cc[t] = __builtin_amdgcn_wmma_f32_16x16x32_bf16(
                            false, a, false, b, (short)0, cc[t], false, false);
            }
        }

        // fold all four 16x16 logit tiles into running (max, sumexp) per row
        #pragma unroll
        for (int r = 0; r < 32; ++r) {
            float val = cc[r >> 3][r & 7];
            float tm = val;
            tm = fmaxf(tm, __shfl_xor(tm, 1, 32));
            tm = fmaxf(tm, __shfl_xor(tm, 2, 32));
            tm = fmaxf(tm, __shfl_xor(tm, 4, 32));
            tm = fmaxf(tm, __shfl_xor(tm, 8, 32));
            float p  = __expf(val - tm);
            float ts = p;
            ts += __shfl_xor(ts, 1, 32);
            ts += __shfl_xor(ts, 2, 32);
            ts += __shfl_xor(ts, 4, 32);
            ts += __shfl_xor(ts, 8, 32);
            float nm = fmaxf(rm[r], tm);
            rs[r] = rs[r] * __expf(rm[r] - nm) + ts * __expf(tm - nm);
            rm[r] = nm;

            int n = mb * 64 + (r >> 3) * 16 + (r & 7) + khalf * 8;
            if (n < N_TOK) {
                int lb = labels[n + 1];
                if (lb == v) atomicAdd(&ws_label[n], val);
            }
        }
    }

    if (col == 0) {
        #pragma unroll
        for (int r = 0; r < 32; ++r) {
            int n = mb * 64 + (r >> 3) * 16 + (r & 7) + khalf * 8;
            size_t idx = ((size_t)n * NVCHUNK + vc) * 2;
            ws_ms[idx]     = rm[r];
            ws_ms[idx + 1] = rs[r];
        }
    }
}

// ---------- fallback pass 1 (on-the-fly cvt, 64 KB LDS A) ----------
__global__ __launch_bounds__(256)
void ce_pass1_kernel(const float* __restrict__ hs,
                     const int*   __restrict__ labels,
                     const float* __restrict__ weight,
                     float* __restrict__ ws_ms,
                     float* __restrict__ ws_label)
{
    __shared__ unsigned int ldsAf[LDS_UINTS];

    const int mb  = blockIdx.x;
    const int vc  = blockIdx.y;
    const int tid = threadIdx.x;

    for (int p = tid; p < LDS_UINTS; p += 256) {
        int ks    = p >> 8;
        int rem   = p & 255;
        int lane  = rem >> 3;
        int j     = rem & 7;
        int row   = lane & 15;
        int khalf = lane >> 4;
        int e     = 2 * j;
        int kl    = e + khalf * 16;
        int k     = ks * 32 + kl;
        int n     = mb * 16 + row;
        float x0 = 0.0f, x1 = 0.0f;
        if (n < N_TOK) {
            x0 = hs[(size_t)n * H_DIM + k];
            x1 = hs[(size_t)n * H_DIM + k + 1];
        }
        ldsAf[p] = pk_bf16x2(x0, x1);
    }
    __syncthreads();

    const int lane   = tid & 31;
    const int waveId = tid >> 5;
    const int col    = lane & 15;
    const int khalf  = lane >> 4;

    float rm[8], rs[8];
    #pragma unroll
    for (int r = 0; r < 8; ++r) { rm[r] = -INFINITY; rs[r] = 0.0f; }

    for (int vt = 0; vt < VCHUNK / 128; ++vt) {
        const int v = vc * VCHUNK + vt * 128 + waveId * 16 + col;
        const float* wrow = weight + (size_t)v * H_DIM + khalf * 16;

        v8f c = {};
        #pragma unroll 2
        for (int ks = 0; ks < H_DIM / 32; ++ks) {
            v16bf a = *reinterpret_cast<const v16bf*>(&ldsAf[ks * 256 + lane * 8]);
            const float4* wp = reinterpret_cast<const float4*>(wrow + ks * 32);
            float4 f0 = wp[0], f1 = wp[1], f2 = wp[2], f3 = wp[3];
            v16bf b;
            b[0]  = (__bf16)f0.x; b[1]  = (__bf16)f0.y; b[2]  = (__bf16)f0.z; b[3]  = (__bf16)f0.w;
            b[4]  = (__bf16)f1.x; b[5]  = (__bf16)f1.y; b[6]  = (__bf16)f1.z; b[7]  = (__bf16)f1.w;
            b[8]  = (__bf16)f2.x; b[9]  = (__bf16)f2.y; b[10] = (__bf16)f2.z; b[11] = (__bf16)f2.w;
            b[12] = (__bf16)f3.x; b[13] = (__bf16)f3.y; b[14] = (__bf16)f3.z; b[15] = (__bf16)f3.w;
            c = __builtin_amdgcn_wmma_f32_16x16x32_bf16(
                    false, a, false, b, (short)0, c, false, false);
        }

        #pragma unroll
        for (int r = 0; r < 8; ++r) {
            float val = c[r];
            float tm = val;
            tm = fmaxf(tm, __shfl_xor(tm, 1, 32));
            tm = fmaxf(tm, __shfl_xor(tm, 2, 32));
            tm = fmaxf(tm, __shfl_xor(tm, 4, 32));
            tm = fmaxf(tm, __shfl_xor(tm, 8, 32));
            float p  = __expf(val - tm);
            float ts = p;
            ts += __shfl_xor(ts, 1, 32);
            ts += __shfl_xor(ts, 2, 32);
            ts += __shfl_xor(ts, 4, 32);
            ts += __shfl_xor(ts, 8, 32);
            float nm = fmaxf(rm[r], tm);
            rs[r] = rs[r] * __expf(rm[r] - nm) + ts * __expf(tm - nm);
            rm[r] = nm;

            int n = mb * 16 + (r + khalf * 8);
            if (n < N_TOK) {
                int lb = labels[n + 1];
                if (lb == v) atomicAdd(&ws_label[n], val);
            }
        }
    }

    if (col == 0) {
        #pragma unroll
        for (int r = 0; r < 8; ++r) {
            int n = mb * 16 + (r + khalf * 8);
            size_t idx = ((size_t)n * NVCHUNK + vc) * 2;
            ws_ms[idx]     = rm[r];
            ws_ms[idx + 1] = rs[r];
        }
    }
}

// ---------- pass 2: deterministic single-block reduction ----------
__global__ __launch_bounds__(256)
void ce_pass2_kernel(const int* __restrict__ labels,
                     const float* __restrict__ ws_ms,
                     const float* __restrict__ ws_label,
                     float* __restrict__ out)
{
    __shared__ float sl[256];
    __shared__ int   sc[256];
    float lossAcc = 0.0f;
    int   cnt     = 0;
    for (int n = threadIdx.x; n < N_TOK; n += 256) {
        int lb = labels[n + 1];
        bool valid = (lb != -100);
        float M = -INFINITY;
        #pragma unroll 5
        for (int j = 0; j < NVCHUNK; ++j)
            M = fmaxf(M, ws_ms[((size_t)n * NVCHUNK + j) * 2]);
        float S = 0.0f;
        #pragma unroll 5
        for (int j = 0; j < NVCHUNK; ++j) {
            float m = ws_ms[((size_t)n * NVCHUNK + j) * 2];
            float s = ws_ms[((size_t)n * NVCHUNK + j) * 2 + 1];
            S += s * __expf(m - M);
        }
        float lse = M + __logf(S);
        float nll = lse - ws_label[n];
        if (valid) { lossAcc += nll; cnt++; }
    }
    sl[threadIdx.x] = lossAcc;
    sc[threadIdx.x] = cnt;
    __syncthreads();
    for (int s = 128; s > 0; s >>= 1) {
        if (threadIdx.x < (unsigned)s) {
            sl[threadIdx.x] += sl[threadIdx.x + s];
            sc[threadIdx.x] += sc[threadIdx.x + s];
        }
        __syncthreads();
    }
    if (threadIdx.x == 0) {
        out[0] = sl[0];
        out[1] = (float)sc[0];
    }
}

extern "C" void kernel_launch(void* const* d_in, const int* in_sizes, int n_in,
                              void* d_out, int out_size, void* d_ws, size_t ws_size,
                              hipStream_t stream) {
    const float* hs     = (const float*)d_in[0];   // (1,2048,2048) fp32
    const int*   labels = (const int*)  d_in[1];   // (1,2048) int32
    const float* weight = (const float*)d_in[2];   // (32000,2048) fp32
    float* out = (float*)d_out;

    char* base = (char*)d_ws;
    float* ws_label = (float*)base;
    float* ws_ms    = (float*)(base + OFF_MS);

    ce_init_kernel<<<1, 256, 0, stream>>>(ws_label);

    if (ws_size >= WS_NEED_FAST) {
        unsigned int*   a_pack = (unsigned int*)(base + OFF_APACK);
        unsigned short* wbf    = (unsigned short*)(base + OFF_WBF);
        // one-time bf16 conversion of weight (fits in 192 MB L2) + A pre-pack
        ce_cvtw_kernel<<<64000, 256, 0, stream>>>(weight, (uint2*)wbf);
        ce_packa_kernel<<<8192, 256, 0, stream>>>(hs, a_pack);
        ce_pass1_fast<<<dim3(MTILES64, NVCHUNK), 256, FAST_LDS_BYTES, stream>>>(
            labels, wbf, a_pack, ws_ms, ws_label);
    } else {
        ce_pass1_kernel<<<dim3(MTILES16, NVCHUNK), 256, 0, stream>>>(
            hs, labels, weight, ws_ms, ws_label);
    }
    ce_pass2_kernel<<<1, 256, 0, stream>>>(labels, ws_ms, ws_label, out);
}
